// ConfidenceBiasedCrossAttention_90529320665516
// MI455X (gfx1250) — matrix operand order repaired and checked by reading, not compiled, not run on hardware
//
#include <hip/hip_runtime.h>

// ---------------- WMMA types (CDNA5 / gfx1250, wave32) ----------------
typedef __attribute__((ext_vector_type(16))) __bf16 v16bf;
typedef __attribute__((ext_vector_type(8)))  float  v8f;

union FragB {
    v16bf v;
    uint4 q[2];
};

static __device__ __forceinline__ v8f wmma_bf16(const FragB& a, const FragB& b, v8f c) {
    // D = A(16x32 bf16) * B(32x16 bf16) + C(16x16 f32)
    return __builtin_amdgcn_wmma_f32_16x16x32_bf16(
        /*neg_a=*/false, a.v, /*neg_b=*/false, b.v,
        /*c_mod=*/(short)0, c, /*reuse_a=*/false, /*reuse_b=*/false);
}

// ---------------- f32 -> bf16 convert ----------------
__global__ void __launch_bounds__(256) cvt_f32_bf16(const float* __restrict__ in,
                                                    __bf16* __restrict__ out, int n) {
    int i = blockIdx.x * blockDim.x + threadIdx.x;
    int stride = gridDim.x * blockDim.x;
    for (; i < n; i += stride) out[i] = (__bf16)in[i];
}

// ---------------- GEMM: Y[M,N] = X[M,K] @ W[N,K]^T + bias[N] ----------------
// One wave -> 16x64 output tile. Register-double-buffered over K (step 32,
// unroll 2) so next step's 10x b128 loads fly while current WMMAs execute.
// mode 0: bf16 out, row-major [M,N]
// mode 1: bf16 out, head-transposed [B,H=16,D=64,Lseg]   (for V)
// mode 2: f32  out, row-major [M,N]                      (final projection)
__global__ void __launch_bounds__(256) gemm_bf16_kernel(
    const __bf16* __restrict__ X, const __bf16* __restrict__ W,
    const float* __restrict__ bias, void* __restrict__ outp,
    int M, int N, int K, int Lseg, int mode)
{
    const int lane   = threadIdx.x & 31;
    const int wid    = blockIdx.x * (blockDim.x >> 5) + (threadIdx.x >> 5);
    const int tilesM = M >> 4;
    const int tilesN = N >> 6;
    if (wid >= tilesM * tilesN) return;
    const int tm = wid % tilesM;
    const int tn = wid / tilesM;
    const int r  = lane & 15;
    const int hi = lane >> 4;
    const int Kq = K >> 3;                     // uint4 (8 bf16) per row

    const uint4* X4 = (const uint4*)X;
    const uint4* W4 = (const uint4*)W;
    const int arow  = (tm * 16 + r) * Kq;      // A: row = query/pixel
    const int wrow0 = (tn * 64 + r) * Kq;      // B: row = output column (W row)

    v8f acc[4] = {};

    auto loadA = [&](FragB& a, int kk) {
        const int kq = kk >> 3;
        a.q[0] = X4[arow + kq + hi * 2];
        a.q[1] = X4[arow + kq + hi * 2 + 1];
    };
    auto loadB = [&](FragB* bf, int kk) {
        const int kq = kk >> 3;
#pragma unroll
        for (int t = 0; t < 4; ++t) {
            const int wr = wrow0 + t * 16 * Kq + kq + hi * 2;
            bf[t].q[0] = W4[wr];
            bf[t].q[1] = W4[wr + 1];
        }
    };
    auto comp = [&](const FragB& a, const FragB* bf) {
#pragma unroll
        for (int t = 0; t < 4; ++t) acc[t] = wmma_bf16(a, bf[t], acc[t]);
    };

    FragB a0, a1, b0[4], b1[4];
    loadA(a0, 0); loadB(b0, 0);
    int kk = 0;
    for (; kk + 64 < K; kk += 64) {
        loadA(a1, kk + 32); loadB(b1, kk + 32);
        __builtin_prefetch(&X4[arow + (kk >> 3) + 16], 0, 3);
        comp(a0, b0);
        loadA(a0, kk + 64); loadB(b0, kk + 64);
        comp(a1, b1);
    }
    loadA(a1, kk + 32); loadB(b1, kk + 32);
    comp(a0, b0);
    comp(a1, b1);

#pragma unroll
    for (int t = 0; t < 4; ++t) {
        const int col = tn * 64 + t * 16 + r;
        const float bv = bias[col];
        if (mode == 1) {
            // [B, H=16, D=64, Lseg]; 8 consecutive l per lane -> packed 16B store
            const int bidx = (tm * 16) / Lseg;
            const int l0   = (tm * 16) % Lseg + hi * 8;
            const int d    = t * 16 + r;       // head = tn (N==1024, 64-wide tiles)
            union { uint4 u; __bf16 e[8]; } pk;
#pragma unroll
            for (int j = 0; j < 8; ++j) pk.e[j] = (__bf16)(acc[t][j] + bv);
            ((uint4*)outp)[((((bidx * 16 + tn) * 64 + d) * Lseg) + l0) >> 3] = pk.u;
        } else if (mode == 2) {
            float* of = (float*)outp;
#pragma unroll
            for (int j = 0; j < 8; ++j)
                of[(tm * 16 + hi * 8 + j) * N + col] = acc[t][j] + bv;
        } else {
            __bf16* ob = (__bf16*)outp;
#pragma unroll
            for (int j = 0; j < 8; ++j)
                ob[(tm * 16 + hi * 8 + j) * N + col] = (__bf16)(acc[t][j] + bv);
        }
    }
}

// ---------------- Flash attention ----------------
// q: [B*Lq, 1024] bf16, k: [B*Lk, 1024] bf16, vT: [B,H,64,Lk] bf16,
// vbias: [B,Lk] f32, ctx: [B*Lq, 1024] bf16.
// One wave: 16 query rows of one head; 128 key-chunks of 32, software-pipelined:
//   V(cur) loads -> S WMMAs (K regs from prev iter) -> K(next) loads ->
//   softmax (overlaps loads) -> P via LDS -> PV WMMAs.
__global__ void __launch_bounds__(128) attn_kernel(
    const __bf16* __restrict__ qb, const __bf16* __restrict__ kb,
    const __bf16* __restrict__ vtb, const float* __restrict__ vbias,
    __bf16* __restrict__ ctx)
{
    __shared__ __attribute__((aligned(16))) __bf16 plds[4][16][32];

    const int warp = threadIdx.x >> 5;
    const int lane = threadIdx.x & 31;
    const int w    = blockIdx.x * 4 + warp;   // 2048 wave-tiles total
    const int b    = w >> 10;                 // 16 heads * 64 qtiles per batch
    const int rem  = w & 1023;
    const int h    = rem >> 6;
    const int qt   = rem & 63;
    const int r    = lane & 15;
    const int hi   = lane >> 4;
    const float scale = 0.125f;               // 1/sqrt(64)

    // ---- load Q A-fragments (d 0..31 and 32..63), row stride = 128 uint4
    const uint4* q4 = (const uint4*)qb;
    const int qrow = (b * 1024 + qt * 16 + r) * 128 + h * 8;
    FragB aq0, aq1;
    aq0.q[0] = q4[qrow + hi * 2];     aq0.q[1] = q4[qrow + hi * 2 + 1];
    aq1.q[0] = q4[qrow + 4 + hi * 2]; aq1.q[1] = q4[qrow + 4 + hi * 2 + 1];

    float m[8], l[8];
#pragma unroll
    for (int j = 0; j < 8; ++j) { m[j] = -1e30f; l[j] = 0.0f; }
    v8f o[4] = {};

    const uint4* k4 = (const uint4*)kb;
    const uint4* v4 = (const uint4*)vtb;
    const uint4* pl = (const uint4*)&plds[warp][0][0];  // 4 uint4 per 32-key row

    auto loadK = [&](FragB* kf, int nb) {
        const int kr0 = (b * 4096 + nb + r) * 128 + h * 8;
        kf[0].q[0] = k4[kr0 + hi * 2];     kf[0].q[1] = k4[kr0 + hi * 2 + 1];
        kf[1].q[0] = k4[kr0 + 4 + hi * 2]; kf[1].q[1] = k4[kr0 + 4 + hi * 2 + 1];
        const int kr1 = (b * 4096 + nb + 16 + r) * 128 + h * 8;
        kf[2].q[0] = k4[kr1 + hi * 2];     kf[2].q[1] = k4[kr1 + hi * 2 + 1];
        kf[3].q[0] = k4[kr1 + 4 + hi * 2]; kf[3].q[1] = k4[kr1 + 4 + hi * 2 + 1];
    };

    auto chunk = [&](const FragB* kf, FragB* kn, int nb, int nbn) {
        // 1) V fragments for this chunk (consumed only after softmax)
        FragB vf[4];
#pragma unroll
        for (int g = 0; g < 4; ++g) {
            const int vr = ((b * 16 + h) * 64 + g * 16 + r) * 512 + (nb >> 3) + hi * 2;
            vf[g].q[0] = v4[vr];
            vf[g].q[1] = v4[vr + 1];
        }
        // 2) S = Q @ K^T (K fragments already resident from previous iteration)
        v8f s0 = {}, s1 = {};
        s0 = wmma_bf16(aq0, kf[0], s0);
        s0 = wmma_bf16(aq1, kf[1], s0);
        s1 = wmma_bf16(aq0, kf[2], s1);
        s1 = wmma_bf16(aq1, kf[3], s1);
        // 3) prefetch next chunk's K fragments (wrapped index keeps it in-bounds)
        loadK(kn, nbn);

        const float bias0 = vbias[b * 4096 + nb + r];
        const float bias1 = vbias[b * 4096 + nb + 16 + r];

        // 4) online softmax; C layout: vgpr j = row (j | j+8), col = lane&15
#pragma unroll
        for (int j = 0; j < 8; ++j) {
            float v0 = s0[j] * scale + bias0;
            float v1 = s1[j] * scale + bias1;
            float mx = fmaxf(v0, v1);
            mx = fmaxf(mx, __shfl_xor(mx, 1, 32));
            mx = fmaxf(mx, __shfl_xor(mx, 2, 32));
            mx = fmaxf(mx, __shfl_xor(mx, 4, 32));
            mx = fmaxf(mx, __shfl_xor(mx, 8, 32));
            const float mn = fmaxf(m[j], mx);
            const float f  = __expf(m[j] - mn);
            const float p0 = __expf(v0 - mn);
            const float p1 = __expf(v1 - mn);
            float ps = p0 + p1;
            ps += __shfl_xor(ps, 1, 32);
            ps += __shfl_xor(ps, 2, 32);
            ps += __shfl_xor(ps, 4, 32);
            ps += __shfl_xor(ps, 8, 32);
            l[j] = l[j] * f + ps;
            m[j] = mn;
            o[0][j] *= f; o[1][j] *= f; o[2][j] *= f; o[3][j] *= f;
            plds[warp][hi * 8 + j][r]      = (__bf16)p0;
            plds[warp][hi * 8 + j][16 + r] = (__bf16)p1;
        }

        asm volatile("s_wait_dscnt 0x0" ::: "memory");

        // 5) P (16x32) as A-fragment from LDS: two ds_load_b128 per lane
        FragB ap;
        ap.q[0] = pl[r * 4 + hi * 2];
        ap.q[1] = pl[r * 4 + hi * 2 + 1];

        // 6) O += P @ V
#pragma unroll
        for (int g = 0; g < 4; ++g) o[g] = wmma_bf16(ap, vf[g], o[g]);
    };

    FragB kfA[4], kfB[4];
    loadK(kfA, 0);
    for (int nb = 0; nb < 4096; nb += 64) {
        chunk(kfA, kfB, nb, nb + 32);
        chunk(kfB, kfA, nb + 32, (nb + 64) & 4095);
    }

    // ---- normalize and store ctx [B*Lq, 1024]
#pragma unroll
    for (int g = 0; g < 4; ++g) {
        const int col = h * 64 + g * 16 + r;
#pragma unroll
        for (int j = 0; j < 8; ++j) {
            const int row = qt * 16 + hi * 8 + j;
            ctx[(b * 1024 + row) * 1024 + col] = (__bf16)(o[g][j] / l[j]);
        }
    }
}

// ---------------- host launch ----------------
extern "C" void kernel_launch(void* const* d_in, const int* in_sizes, int n_in,
                              void* d_out, int out_size, void* d_ws, size_t ws_size,
                              hipStream_t stream) {
    const float* Q    = (const float*)d_in[0];
    const float* K_in = (const float*)d_in[1];
    const float* V_in = (const float*)d_in[2];
    const float* Vb   = (const float*)d_in[3];
    const float* Wq_w = (const float*)d_in[4];
    const float* Wq_b = (const float*)d_in[5];
    const float* Wk_w = (const float*)d_in[6];
    const float* Wk_b = (const float*)d_in[7];
    const float* Wv_w = (const float*)d_in[8];
    const float* Wv_b = (const float*)d_in[9];
    const float* Wo_w = (const float*)d_in[10];
    const float* Wo_b = (const float*)d_in[11];

    const int nQ = 2 * 1024 * 1024;   // 2,097,152
    const int nK = 2 * 4096 * 1024;   // 8,388,608
    const int nW = 1024 * 1024;

    char* p = (char*)d_ws;
    auto alloc = [&](size_t bytes) {
        char* ret = p;
        p += (bytes + 255) & ~(size_t)255;
        return ret;
    };
    __bf16* Xq   = (__bf16*)alloc((size_t)nQ * 2);
    __bf16* Xk   = (__bf16*)alloc((size_t)nK * 2);
    __bf16* Xv   = (__bf16*)alloc((size_t)nK * 2);
    __bf16* Wqc  = (__bf16*)alloc((size_t)nW * 2);
    __bf16* Wkc  = (__bf16*)alloc((size_t)nW * 2);
    __bf16* Wvc  = (__bf16*)alloc((size_t)nW * 2);
    __bf16* Woc  = (__bf16*)alloc((size_t)nW * 2);
    __bf16* qbuf = (__bf16*)alloc((size_t)nQ * 2);
    __bf16* kbuf = (__bf16*)alloc((size_t)nK * 2);
    __bf16* vtbf = (__bf16*)alloc((size_t)nK * 2);
    __bf16* ctx  = (__bf16*)alloc((size_t)nQ * 2);

    cvt_f32_bf16<<<2048, 256, 0, stream>>>(Q, Xq, nQ);
    cvt_f32_bf16<<<2048, 256, 0, stream>>>(K_in, Xk, nK);
    cvt_f32_bf16<<<2048, 256, 0, stream>>>(V_in, Xv, nK);
    cvt_f32_bf16<<<1024, 256, 0, stream>>>(Wq_w, Wqc, nW);
    cvt_f32_bf16<<<1024, 256, 0, stream>>>(Wk_w, Wkc, nW);
    cvt_f32_bf16<<<1024, 256, 0, stream>>>(Wv_w, Wvc, nW);
    cvt_f32_bf16<<<1024, 256, 0, stream>>>(Wo_w, Woc, nW);

    // q = Q @ Wq^T + bq  -> [2048,1024] bf16
    gemm_bf16_kernel<<<256, 256, 0, stream>>>(Xq, Wqc, Wq_b, qbuf, 2048, 1024, 1024, 0, 0);
    // k = K @ Wk^T + bk  -> [8192,1024] bf16
    gemm_bf16_kernel<<<1024, 256, 0, stream>>>(Xk, Wkc, Wk_b, kbuf, 8192, 1024, 1024, 0, 0);
    // v = V @ Wv^T + bv  -> head-transposed [2,16,64,4096] bf16
    gemm_bf16_kernel<<<1024, 256, 0, stream>>>(Xv, Wvc, Wv_b, vtbf, 8192, 1024, 1024, 4096, 1);
    // flash attention -> ctx [2048,1024] bf16
    attn_kernel<<<512, 128, 0, stream>>>(qbuf, kbuf, vtbf, Vb, ctx);
    // out = ctx @ Wo^T + bo -> f32 d_out
    gemm_bf16_kernel<<<256, 256, 0, stream>>>(ctx, Woc, Wo_b, d_out, 2048, 1024, 1024, 0, 2);
}
